// RSFConv2d_61435212202243
// MI455X (gfx1250) — matrix operand order repaired and checked by reading, not compiled, not run on hardware
//
#include <hip/hip_runtime.h>
#include <hip/hip_bf16.h>

typedef __attribute__((ext_vector_type(16))) __bf16 v16bf;
typedef __attribute__((ext_vector_type(8)))  float  v8f;

#define CIN    32
#define COUT   32
#define HH     512
#define WW     512
#define TILE_W 64
#define TILE_H 4
#define TROWS  (TILE_H + 2)       // 6 input rows
#define TCOLS  (TILE_W + 2)       // 66 input cols
#define CPADH  40                 // ci stride (bf16 units): 80 B, 16B-aligned, bank-tiling
#define NFRAG  (9 * 2 * 32 * 8)   // taps * co-tiles * lanes * words (u32)

__device__ __forceinline__ unsigned bf16_rne_bits(float f) {
  unsigned u = __float_as_uint(f);
  return (u + 0x7FFFu + ((u >> 16) & 1u)) >> 16;
}

// ---------------------------------------------------------------------------
// Kernel 1: build equivariant 3x3 kernel and pre-pack WMMA A-fragments
// (bf16 hi/lo split), in contiguous k order (lane half h, element s -> k = 16h + s), same as B.
// ---------------------------------------------------------------------------
__global__ void prep_weights(const float* __restrict__ fw,
                             unsigned* __restrict__ afrag_hi,
                             unsigned* __restrict__ afrag_lo) {
  __shared__ float kern[COUT * CIN * 9];   // [co][ci][tap], 36 KB
  const int tid = threadIdx.x;

  const float ct3[3] = {1.f, -0.5f, -0.5f};
  const float st3[3] = {0.f, 0.8660254037844386f, -0.8660254037844386f};
  const float crd[3] = {-2.f / 3.f, 0.f, 2.f / 3.f};

  for (int pair = tid; pair < COUT * CIN; pair += blockDim.x) {
    const float* w = fw + pair * (3 * 2 * 2);   // (3,2,2) per (o,i)
    float Cr[3][3], Ci[3][3];
    for (int u = 0; u < 3; ++u)
      for (int v = 0; v < 2; ++v) {
        Cr[u][v] = w[(u * 2 + v) * 2 + 0];
        Ci[u][v] = w[(u * 2 + v) * 2 + 1];
      }
    for (int u = 0; u < 3; ++u) {
      int um = (3 - u) % 3;
      Cr[u][2] =  Cr[um][1];
      Ci[u][2] = -Ci[um][1];
    }
    float base[3][3];
    for (int y = 0; y < 3; ++y)
      for (int x = 0; x < 3; ++x) {
        float s = 0.f;
        for (int u = 0; u < 3; ++u)
          for (int v = 0; v < 3; ++v) {
            int k = (u * y + v * x) % 3;
            s += Cr[u][v] * ct3[k] - Ci[u][v] * st3[k];
          }
        base[y][x] = s * (1.f / 9.f);
      }
    float acc[3][3] = {};
    for (int j = 0; j < 32; ++j) {
      int r = j >> 2, sidx = j & 3;
      float scl = 1.f;
      for (int q = 0; q < sidx; ++q) scl *= 1.25f;
      float th = (float)r * 0.78539816339744831f;
      float c = cosf(th) * scl, s = sinf(th) * scl;
      for (int oy = 0; oy < 3; ++oy)
        for (int ox = 0; ox < 3; ++ox) {
          float gxo = crd[ox], gyo = crd[oy];
          float gx = c * gxo - s * gyo;
          float gy = s * gxo + c * gyo;
          float ix = ((gx + 1.f) * 3.f - 1.f) * 0.5f;
          float iy = ((gy + 1.f) * 3.f - 1.f) * 0.5f;
          float fx0 = floorf(ix), fy0 = floorf(iy);
          int x0 = (int)fx0, y0 = (int)fy0;
          float wx1 = ix - fx0, wy1 = iy - fy0;
          float wx0 = 1.f - wx1, wy0 = 1.f - wy1;
          auto tap = [&](int yy, int xx) -> float {
            if (xx < 0 || xx > 2 || yy < 0 || yy > 2) return 0.f;
            return base[yy][xx];
          };
          acc[oy][ox] += tap(y0, x0)     * wy0 * wx0
                       + tap(y0, x0 + 1) * wy0 * wx1
                       + tap(y0 + 1, x0) * wy1 * wx0
                       + tap(y0 + 1, x0 + 1) * wy1 * wx1;
        }
    }
    for (int oy = 0; oy < 3; ++oy)
      for (int ox = 0; ox < 3; ++ox)
        kern[pair * 9 + oy * 3 + ox] = acc[oy][ox] * (1.f / 32.f);
  }
  __syncthreads();

  // Pack A fragments: idx = ((t*2 + ct)*32 + lane)*8 + v
  for (int idx = tid; idx < NFRAG; idx += blockDim.x) {
    int v  = idx & 7;
    int L  = (idx >> 3) & 31;
    int ct = (idx >> 8) & 1;
    int t  = idx >> 9;
    int m  = L & 15;
    int co = ct * 16 + m;
    int kb = 2 * v + ((L < 16) ? 0 : 16);
    float f0 = kern[(co * CIN + kb) * 9 + t];
    float f1 = kern[(co * CIN + kb + 1) * 9 + t];
    unsigned h0 = bf16_rne_bits(f0), h1 = bf16_rne_bits(f1);
    float r0 = f0 - __uint_as_float(h0 << 16);
    float r1 = f1 - __uint_as_float(h1 << 16);
    afrag_hi[idx] = h0 | (h1 << 16);
    afrag_lo[idx] = bf16_rne_bits(r0) | (bf16_rne_bits(r1) << 16);
  }
}

// ---------------------------------------------------------------------------
// Kernel 2: 3x3 conv as implicit GEMM on v_wmma_f32_16x16x32_bf16.
// 4 output rows x 64 pixels per block; bf16 hi/lo tiles built once in LDS;
// A-fragment table DMA'd to LDS with global_load_async_to_lds_b128.
// ---------------------------------------------------------------------------
__global__ __launch_bounds__(256) void conv_wmma(
    const float* __restrict__ x, const unsigned* __restrict__ afrag,
    float* __restrict__ out) {
  __shared__ __align__(16) unsigned short tileH[TROWS * TCOLS * CPADH]; // 31680 B
  __shared__ __align__(16) unsigned short tileL[TROWS * TCOLS * CPADH]; // 31680 B
  __shared__ __align__(16) unsigned lds_af[2 * NFRAG];                  // 36864 B

  const int tid  = threadIdx.x;
  const int lane = tid & 31;
  const int wv   = tid >> 5;
  const int npx  = lane & 15;
  const int hf   = lane >> 4;
  const int wbase = blockIdx.x * TILE_W;
  const int hbase = blockIdx.y * TILE_H;
  const int n = blockIdx.z;

  // --- Phase 1: kick off async DMA of the A-fragment table into LDS -------
  {
    unsigned lds_base = (unsigned)(size_t)(&lds_af[0]);
    const uint4* g = (const uint4*)afrag;
    for (int i = tid; i < (2 * NFRAG) / 4; i += 256) {
      unsigned lds_off = lds_base + (unsigned)i * 16u;
      const uint4* gp = g + i;
      asm volatile("global_load_async_to_lds_b128 %0, %1, off"
                   :: "v"(lds_off), "v"(gp) : "memory");
    }
  }

  // --- Phase 2: stage input halo tile, f32 -> bf16 hi/lo, [row][col][ci] --
  const float* xn = x + (size_t)n * (CIN * HH * WW);
  for (int idx = tid; idx < TROWS * TCOLS * CIN; idx += 256) {
    int col = idx % TCOLS;
    int t2  = idx / TCOLS;
    int y   = t2 % TROWS;
    int ci  = t2 / TROWS;
    int gy = hbase + y - 1;
    int gx = wbase + col - 1;
    float v = 0.f;
    if ((unsigned)gy < (unsigned)HH && (unsigned)gx < (unsigned)WW)
      v = xn[((size_t)ci * HH + gy) * WW + gx];
    unsigned hb = bf16_rne_bits(v);
    float lo = v - __uint_as_float(hb << 16);
    unsigned lb = bf16_rne_bits(lo);
    int off = (y * TCOLS + col) * CPADH + ci;
    tileH[off] = (unsigned short)hb;
    tileL[off] = (unsigned short)lb;
  }

  asm volatile("s_wait_asynccnt 0x0" ::: "memory");
  __syncthreads();

  // --- Phase 3: implicit GEMM, 2 pixel groups x 2 co-tiles per wave -------
  v8f acc[2][2] = {};
  const int r   = wv >> 1;          // output row within tile (0..3)
  const int cgb = (wv & 1) * 2;     // first 16-pixel column group (0 or 2)
  const int cib = hf * 16;          // B layout: lanes 0-15 K=0..15, 16-31 K=16..31

  union U { uint4 q[2]; v16bf b; };

#pragma unroll
  for (int t = 0; t < 9; ++t) {
    const int kh = t / 3, kw = t % 3;

    U a0h, a0l, a1h, a1l;
    {
      const uint4* p;
      p = (const uint4*)&lds_af[(((t * 2 + 0) * 32 + lane) * 8)];
      a0h.q[0] = p[0]; a0h.q[1] = p[1];
      p = (const uint4*)&lds_af[NFRAG + (((t * 2 + 0) * 32 + lane) * 8)];
      a0l.q[0] = p[0]; a0l.q[1] = p[1];
      p = (const uint4*)&lds_af[(((t * 2 + 1) * 32 + lane) * 8)];
      a1h.q[0] = p[0]; a1h.q[1] = p[1];
      p = (const uint4*)&lds_af[NFRAG + (((t * 2 + 1) * 32 + lane) * 8)];
      a1l.q[0] = p[0]; a1l.q[1] = p[1];
    }

#pragma unroll
    for (int g = 0; g < 2; ++g) {
      const int tc  = (cgb + g) * 16 + npx + kw;
      const int off = ((r + kh) * TCOLS + tc) * CPADH + cib;
      U bh, bl;
      const uint4* pb = (const uint4*)&tileH[off];
      bh.q[0] = pb[0]; bh.q[1] = pb[1];
      const uint4* pl = (const uint4*)&tileL[off];
      bl.q[0] = pl[0]; bl.q[1] = pl[1];

      acc[g][0] = __builtin_amdgcn_wmma_f32_16x16x32_bf16(false, a0h.b, false, bh.b, (short)0, acc[g][0], false, false);
      acc[g][0] = __builtin_amdgcn_wmma_f32_16x16x32_bf16(false, a0h.b, false, bl.b, (short)0, acc[g][0], false, false);
      acc[g][0] = __builtin_amdgcn_wmma_f32_16x16x32_bf16(false, a0l.b, false, bh.b, (short)0, acc[g][0], false, false);
      acc[g][1] = __builtin_amdgcn_wmma_f32_16x16x32_bf16(false, a1h.b, false, bh.b, (short)0, acc[g][1], false, false);
      acc[g][1] = __builtin_amdgcn_wmma_f32_16x16x32_bf16(false, a1h.b, false, bl.b, (short)0, acc[g][1], false, false);
      acc[g][1] = __builtin_amdgcn_wmma_f32_16x16x32_bf16(false, a1l.b, false, bh.b, (short)0, acc[g][1], false, false);
    }
  }

  // --- Phase 4: store. C/D layout: lanes 0-15 VGPR r -> M=r; 16-31 -> M=8+r
#pragma unroll
  for (int g = 0; g < 2; ++g) {
    float* outp = out + (size_t)n * (COUT * HH * WW)
                + (size_t)(hbase + r) * WW
                + (wbase + (cgb + g) * 16 + npx);
#pragma unroll
    for (int rr = 0; rr < 8; ++rr) {
      int m = hf * 8 + rr;
      outp[(size_t)m * (HH * WW)]        = acc[g][0][rr];
      outp[(size_t)(m + 16) * (HH * WW)] = acc[g][1][rr];
    }
  }
}

// ---------------------------------------------------------------------------
extern "C" void kernel_launch(void* const* d_in, const int* in_sizes, int n_in,
                              void* d_out, int out_size, void* d_ws, size_t ws_size,
                              hipStream_t stream) {
  const float* x  = (const float*)d_in[0];
  const float* fw = (const float*)d_in[1];
  float* out = (float*)d_out;

  unsigned* afhi = (unsigned*)d_ws;
  unsigned* aflo = afhi + NFRAG;   // contiguous: [hi | lo]

  prep_weights<<<1, 256, 0, stream>>>(fw, afhi, aflo);

  const int nb = in_sizes[0] / (CIN * HH * WW);   // batch = 8
  dim3 grid(WW / TILE_W, HH / TILE_H, nb);
  conv_wmma<<<grid, dim3(256), 0, stream>>>(x, afhi, out);
}
